// Net_83794811945606
// MI455X (gfx1250) — matrix operand order repaired and checked by reading, not compiled, not run on hardware
//
#include <hip/hip_runtime.h>
#include <math.h>
#include <stdint.h>

// ---------------------------------------------------------------------------
// Types for CDNA5 WMMA
// ---------------------------------------------------------------------------
typedef __attribute__((ext_vector_type(16))) __bf16 v16bf;
typedef __attribute__((ext_vector_type(8)))  float  v8f;

union AFrag { v16bf v; uint4 q[2]; };

__device__ __forceinline__ unsigned short f2bf(float f) {
    unsigned int u = __float_as_uint(f);
    unsigned int r = u + 0x7FFFu + ((u >> 16) & 1u);   // round-to-nearest-even
    return (unsigned short)(r >> 16);
}

// CDNA5 async global->LDS copy (ASYNCcnt-tracked), 16 bytes per lane.
__device__ __forceinline__ void async_load_b128(unsigned lds_addr,
                                                const unsigned short* gptr) {
    unsigned long long ga = (unsigned long long)(uintptr_t)gptr;
    asm volatile("global_load_async_to_lds_b128 %0, %1, off"
                 :: "v"(lds_addr), "v"(ga) : "memory");
}
__device__ __forceinline__ void wait_async0() {
    asm volatile("s_wait_asynccnt 0x0" ::: "memory");
}

// ---------------------------------------------------------------------------
// GEMM: C[M,Nn] = A[M,K](bf16) * Bt[Npad,K](bf16, pre-transposed) (+bias[Nn])
// Block tile 128x128x32, 8 waves, wave tile 32x64 (2x4 WMMA frags),
// double-buffered LDS filled with async-to-LDS copies.
// ---------------------------------------------------------------------------
__global__ __launch_bounds__(256)
void gemm_bf16_wmma(const unsigned short* __restrict__ A,   // [M][K]
                    const unsigned short* __restrict__ Bt,  // [Npad][K]
                    float* __restrict__ C,
                    const float* __restrict__ bias,
                    int M, int Nn, int K)
{
    __shared__ unsigned short sA[2][128 * 32];   // [m][k]
    __shared__ unsigned short sB[2][128 * 32];   // [n][k]

    const int tid  = threadIdx.x;
    const int lane = tid & 31;
    const int wave = tid >> 5;
    const int wm   = wave >> 1;              // 0..3  (rows of 32)
    const int wn   = wave & 1;               // 0..1  (cols of 64)
    const int blockM = blockIdx.y * 128;
    const int blockN = blockIdx.x * 128;
    const bool fullM = (blockM + 128) <= M;  // uniform: no M-tail in this block

    // staging assignment: each thread owns 16 contiguous elements of one row
    const int row = tid >> 1;                // 0..127
    const int col = (tid & 1) * 16;          // 0 or 16
    const int gm  = blockM + row;

    const unsigned short* gA = A  + (size_t)gm * K + col;
    const unsigned short* gB = Bt + (size_t)(blockN + row) * K + col;

    const unsigned sAbase = (unsigned)(uintptr_t)&sA[0][0];
    const unsigned sBbase = (unsigned)(uintptr_t)&sB[0][0];
    const unsigned stOff  = (unsigned)((row * 32 + col) * 2);
    const unsigned bufBytes = 128 * 32 * 2;

    auto stage = [&](int b, int kt) {
        const unsigned a0 = sAbase + (unsigned)b * bufBytes + stOff;
        const unsigned b0 = sBbase + (unsigned)b * bufBytes + stOff;
        if (fullM) {
            async_load_b128(a0,      gA + kt);
            async_load_b128(a0 + 16, gA + kt + 8);
        } else if (gm < M) {
            async_load_b128(a0,      gA + kt);
            async_load_b128(a0 + 16, gA + kt + 8);
        }
        async_load_b128(b0,      gB + kt);
        async_load_b128(b0 + 16, gB + kt + 8);
    };

    v8f acc[2][4];
#pragma unroll
    for (int i = 0; i < 2; ++i)
#pragma unroll
        for (int j = 0; j < 4; ++j)
            acc[i][j] = v8f{0.f,0.f,0.f,0.f,0.f,0.f,0.f,0.f};

    const int nk = K >> 5;
    stage(0, 0);

    const int mlo = lane & 15;
    const int kbA = (lane < 16) ? 0 : 8;     // A frag: lanes16-31 hold K 8..15/24..31
    const int kbB = (lane < 16) ? 0 : 8;     // B frag: lanes16-31 hold K 8..15/24..31

    for (int it = 0; it < nk; ++it) {
        wait_async0();
        __syncthreads();
        if (it + 1 < nk) stage((it + 1) & 1, (it + 1) * 32);

        const unsigned short* cA = sA[it & 1];
        const unsigned short* cB = sB[it & 1];

        // Fragments: each is two aligned 16B LDS chunks -> ds_load_b128
        AFrag aF[2], bF[4];
#pragma unroll
        for (int fa = 0; fa < 2; ++fa) {
            const unsigned short* base = cA + (wm * 32 + fa * 16 + mlo) * 32;
            aF[fa].q[0] = *(const uint4*)(base + kbA);        // K kb..kb+7
            aF[fa].q[1] = *(const uint4*)(base + 16 + kbA);   // K 16+kb..16+kb+7
        }
#pragma unroll
        for (int fb = 0; fb < 4; ++fb) {
            const unsigned short* base = cB + (wn * 64 + fb * 16 + mlo) * 32;
            bF[fb].q[0] = *(const uint4*)(base + kbB);        // K kb..kb+7
            bF[fb].q[1] = *(const uint4*)(base + 16 + kbB);   // K 16+kb..16+kb+7
        }
#pragma unroll
        for (int fa = 0; fa < 2; ++fa)
#pragma unroll
            for (int fb = 0; fb < 4; ++fb)
                acc[fa][fb] = __builtin_amdgcn_wmma_f32_16x16x32_bf16(
                    false, aF[fa].v, false, bF[fb].v,
                    (short)0, acc[fa][fb], false, false);
    }

    // ---- store D (C/D layout: VGPR r -> M = r (+8 for lanes 16..31)) ----
    const int mhi = (lane < 16) ? 0 : 8;
#pragma unroll
    for (int fa = 0; fa < 2; ++fa) {
#pragma unroll
        for (int fb = 0; fb < 4; ++fb) {
            const int gn = blockN + wn * 64 + fb * 16 + mlo;
            if (gn >= Nn) continue;
            const float bv = bias ? bias[gn] : 0.0f;
#pragma unroll
            for (int r = 0; r < 8; ++r) {
                const int gmm = blockM + wm * 32 + fa * 16 + mhi + r;
                if (gmm < M) C[(size_t)gmm * Nn + gn] = acc[fa][fb][r] + bv;
            }
        }
    }
}

// ---------------------------------------------------------------------------
// Elementwise helpers
// ---------------------------------------------------------------------------
__global__ void f32_to_bf16_k(const float* __restrict__ in,
                              unsigned short* __restrict__ out, int n) {
    int i = blockIdx.x * blockDim.x + threadIdx.x;
    if (i < n) out[i] = f2bf(in[i]);
}

// W[K][N] fp32 -> out[Npad][K] bf16 (transposed + zero-padded rows)
__global__ void cvt_w_transpose_k(const float* __restrict__ W,
                                  unsigned short* __restrict__ out,
                                  int K, int N, int Npad) {
    int i = blockIdx.x * blockDim.x + threadIdx.x;
    if (i >= Npad * K) return;
    int n = i / K, k = i - n * K;
    float v = (n < N) ? W[(size_t)k * N + n] : 0.0f;
    out[i] = f2bf(v);
}

__global__ void fill_f32_k(float* __restrict__ p, int n, float v) {
    int i = blockIdx.x * blockDim.x + threadIdx.x;
    if (i < n) p[i] = v;
}

__global__ void fix_emax_k(float* __restrict__ p, int n) {
    int i = blockIdx.x * blockDim.x + threadIdx.x;
    if (i < n) { float v = p[i]; if (!(v >= -3.0e38f)) p[i] = 0.0f; }
}

// x = elu(x + bias), also emit bf16 copy for the next GEMM
__global__ void bias_elu_bf16_k(float* __restrict__ X, const float* __restrict__ bias,
                                int total, int Cdim, unsigned short* __restrict__ xb) {
    int i = blockIdx.x * blockDim.x + threadIdx.x;
    if (i >= total) return;
    float v = X[i] + bias[i % Cdim];
    v = (v > 0.0f) ? v : (expf(v) - 1.0f);
    X[i] = v;
    xb[i] = f2bf(v);
}

__global__ void add_bias_k(float* __restrict__ X, const float* __restrict__ bias,
                           int total, int Cdim) {
    int i = blockIdx.x * blockDim.x + threadIdx.x;
    if (i < total) X[i] += bias[i % Cdim];
}

// ---------------------------------------------------------------------------
// Attention: per-(node,head) dot products, one wave32 per (n,h)
// ---------------------------------------------------------------------------
__global__ __launch_bounds__(256)
void att_scores_k(const float* __restrict__ Hf,
                  const float* __restrict__ attS, const float* __restrict__ attD,
                  float* __restrict__ asrc, float* __restrict__ adst,
                  int N, int heads, int C) {
    int gw   = (blockIdx.x * blockDim.x + threadIdx.x) >> 5;
    int lane = threadIdx.x & 31;
    if (gw >= N * heads) return;
    int n = gw / heads, h = gw - n * heads;
    const float* hp = Hf + (size_t)n * heads * C + (size_t)h * C;
    float ss = 0.f, sd = 0.f;
    for (int c = lane; c < C; c += 32) {
        float v = hp[c];
        ss += v * attS[h * C + c];
        sd += v * attD[h * C + c];
    }
#pragma unroll
    for (int off = 16; off; off >>= 1) {
        ss += __shfl_down(ss, off, 32);
        sd += __shfl_down(sd, off, 32);
    }
    if (lane == 0) { asrc[gw] = ss; adst[gw] = sd; }
}

__device__ __forceinline__ void atomicMaxFloat(float* addr, float v) {
    if (v >= 0.0f) atomicMax((int*)addr, __float_as_int(v));
    else           atomicMin((unsigned int*)addr, __float_as_uint(v));
}

__device__ __forceinline__ void edge_sd(const int* ei, int e, int E, int& s, int& d) {
    if (e < E) { s = ei[e]; d = ei[E + e]; } else { s = d = e - E; }
}

__global__ void edge_max_k(const float* __restrict__ asrc, const float* __restrict__ adst,
                           const int* __restrict__ ei, int E, int ET, int heads,
                           float* __restrict__ emax) {
    int i = blockIdx.x * blockDim.x + threadIdx.x;
    if (i >= ET * heads) return;
    int e = i / heads, h = i - e * heads, s, d;
    edge_sd(ei, e, E, s, d);
    float sc = asrc[s * heads + h] + adst[d * heads + h];
    sc = (sc > 0.0f) ? sc : 0.2f * sc;
    atomicMaxFloat(&emax[d * heads + h], sc);
}

__global__ void edge_den_k(const float* __restrict__ asrc, const float* __restrict__ adst,
                           const float* __restrict__ emax, const int* __restrict__ ei,
                           int E, int ET, int heads, float* __restrict__ deno) {
    int i = blockIdx.x * blockDim.x + threadIdx.x;
    if (i >= ET * heads) return;
    int e = i / heads, h = i - e * heads, s, d;
    edge_sd(ei, e, E, s, d);
    float sc = asrc[s * heads + h] + adst[d * heads + h];
    sc = (sc > 0.0f) ? sc : 0.2f * sc;
    atomicAdd(&deno[d * heads + h], expf(sc - emax[d * heads + h]));
}

// One block per edge: scatter alpha * h[src] into out[dst]
__global__ __launch_bounds__(256)
void gat_scatter_k(const float* __restrict__ Hf,
                   const float* __restrict__ asrc, const float* __restrict__ adst,
                   const float* __restrict__ emax, const float* __restrict__ deno,
                   const int* __restrict__ ei, int E, int heads, int C,
                   float* __restrict__ out, int outStride, int concat, float scale) {
    __shared__ float s_alpha[8];
    int e = blockIdx.x, s, d;
    edge_sd(ei, e, E, s, d);
    int tid = threadIdx.x;
    if (tid < heads) {
        float sc = asrc[s * heads + tid] + adst[d * heads + tid];
        sc = (sc > 0.0f) ? sc : 0.2f * sc;
        float a = expf(sc - emax[d * heads + tid]) / deno[d * heads + tid];
        s_alpha[tid] = a * scale;
    }
    __syncthreads();
    const int HC = heads * C;
    const float* hs = Hf + (size_t)s * HC;
    float* od = out + (size_t)d * outStride;
    for (int ch = tid; ch < HC; ch += blockDim.x) {
        int h = ch / C;
        int oc = concat ? ch : (ch - h * C);
        atomicAdd(&od[oc], s_alpha[h] * hs[ch]);
    }
}

// ---------------------------------------------------------------------------
// Host orchestration
// ---------------------------------------------------------------------------
static inline int padN(int n) { return (n + 127) & ~127; }

static inline void gemm(const unsigned short* A, const unsigned short* Bt, float* C,
                        const float* bias, int M, int Nn, int K, hipStream_t s) {
    dim3 g(padN(Nn) / 128, (M + 127) / 128);
    gemm_bf16_wmma<<<g, 256, 0, s>>>(A, Bt, C, bias, M, Nn, K);
}

static inline void cvt(const float* in, unsigned short* out, int n, hipStream_t s) {
    f32_to_bf16_k<<<(n + 255) / 256, 256, 0, s>>>(in, out, n);
}

static inline void cvtW(const float* W, unsigned short* out, int K, int N, hipStream_t s) {
    int tot = padN(N) * K;
    cvt_w_transpose_k<<<(tot + 255) / 256, 256, 0, s>>>(W, out, K, N, padN(N));
}

static inline void gat_edges(const float* Hf, const float* attS, const float* attD,
                             const int* ei, float* asrc, float* adst, float* emax,
                             float* deno, float* out, int outStride, int concat,
                             float scale, int N, int E, int ET, int heads, int C,
                             hipStream_t s) {
    const int nh = N * heads;
    att_scores_k<<<(nh * 32 + 255) / 256, 256, 0, s>>>(Hf, attS, attD, asrc, adst, N, heads, C);
    fill_f32_k<<<(nh + 255) / 256, 256, 0, s>>>(emax, nh, -__builtin_huge_valf());
    fill_f32_k<<<(nh + 255) / 256, 256, 0, s>>>(deno, nh, 0.0f);
    const int cnt = ET * heads;
    edge_max_k<<<(cnt + 255) / 256, 256, 0, s>>>(asrc, adst, ei, E, ET, heads, emax);
    fix_emax_k<<<(nh + 255) / 256, 256, 0, s>>>(emax, nh);
    edge_den_k<<<(cnt + 255) / 256, 256, 0, s>>>(asrc, adst, emax, ei, E, ET, heads, deno);
    gat_scatter_k<<<ET, 256, 0, s>>>(Hf, asrc, adst, emax, deno, ei, E, heads, C,
                                     out, outStride, concat, scale);
}

extern "C" void kernel_launch(void* const* d_in, const int* in_sizes, int n_in,
                              void* d_out, int out_size, void* d_ws, size_t ws_size,
                              hipStream_t stream) {
    const int N = 50000, E = 400000, NC = 121;
    const int ET = E + N;

    const float* x    = (const float*)d_in[0];
    const int*   ei   = (const int*)  d_in[1];
    const float* mapW = (const float*)d_in[2];
    const float* mapB = (const float*)d_in[3];
    const float* c1W  = (const float*)d_in[4];
    const float* c1aS = (const float*)d_in[5];
    const float* c1aD = (const float*)d_in[6];
    const float* c1B  = (const float*)d_in[7];
    const float* l1W  = (const float*)d_in[8];
    const float* l1B  = (const float*)d_in[9];
    const float* c2W  = (const float*)d_in[10];
    const float* c2aS = (const float*)d_in[11];
    const float* c2aD = (const float*)d_in[12];
    const float* c2B  = (const float*)d_in[13];
    const float* l2W  = (const float*)d_in[14];
    const float* l2B  = (const float*)d_in[15];
    const float* c3W  = (const float*)d_in[16];
    const float* c3aS = (const float*)d_in[17];
    const float* c3aD = (const float*)d_in[18];
    const float* c3B  = (const float*)d_in[19];
    const float* l3W  = (const float*)d_in[20];
    const float* l3B  = (const float*)d_in[21];
    float* out = (float*)d_out;

    // workspace layout
    char* ws = (char*)d_ws;
    unsigned short* Abf = (unsigned short*)(ws);                 // N*1024 bf16
    unsigned short* Wbf = (unsigned short*)(ws + 102400000ULL);  // 1024*1024 bf16 (padded, transposed)
    float* Hf   = (float*)(ws + 104497152ULL);                   // N*1024 f32
    float* S    = (float*)(ws + 309297152ULL);                   // N*1024 f32
    float* asrc = (float*)(ws + 514097152ULL);                   // N*6
    float* adst = asrc + (size_t)N * 6;
    float* emax = adst + (size_t)N * 6;
    float* deno = emax + (size_t)N * 6;

    // ---- mapping: x0 = x @ mapW + mapB ----
    cvt(x, Abf, N * 128, stream);
    cvtW(mapW, Wbf, 128, 128, stream);
    gemm(Abf, Wbf, S, mapB, N, 128, 128, stream);
    cvt(S, Abf, N * 128, stream);                       // x0 -> bf16

    // ---- layer 1: GATConv(128 -> 4x256) + lin1 skip ----
    cvtW(c1W, Wbf, 128, 1024, stream);
    gemm(Abf, Wbf, Hf, nullptr, N, 1024, 128, stream);  // h1
    cvtW(l1W, Wbf, 128, 1024, stream);
    gemm(Abf, Wbf, S, l1B, N, 1024, 128, stream);       // skip1 (accumulator)
    gat_edges(Hf, c1aS, c1aD, ei, asrc, adst, emax, deno,
              S, 1024, /*concat=*/1, 1.0f, N, E, ET, 4, 256, stream);
    bias_elu_bf16_k<<<((N * 1024) + 255) / 256, 256, 0, stream>>>(S, c1B, N * 1024, 1024, Abf);

    // ---- layer 2: GATConv(1024 -> 4x256) + lin2 skip ----
    cvtW(c2W, Wbf, 1024, 1024, stream);
    gemm(Abf, Wbf, Hf, nullptr, N, 1024, 1024, stream); // h2
    cvtW(l2W, Wbf, 1024, 1024, stream);
    gemm(Abf, Wbf, S, l2B, N, 1024, 1024, stream);      // skip2
    gat_edges(Hf, c2aS, c2aD, ei, asrc, adst, emax, deno,
              S, 1024, /*concat=*/1, 1.0f, N, E, ET, 4, 256, stream);
    bias_elu_bf16_k<<<((N * 1024) + 255) / 256, 256, 0, stream>>>(S, c2B, N * 1024, 1024, Abf);

    // ---- layer 3: GATConv(1024 -> 6x121, mean) + lin3 skip -> d_out ----
    cvtW(c3W, Wbf, 1024, 726, stream);
    gemm(Abf, Wbf, Hf, nullptr, N, 726, 1024, stream);  // h3
    cvtW(l3W, Wbf, 1024, 121, stream);
    gemm(Abf, Wbf, out, l3B, N, 121, 1024, stream);     // out = skip3 + l3B
    add_bias_k<<<((N * NC) + 255) / 256, 256, 0, stream>>>(out, c3B, N * NC, NC);
    gat_edges(Hf, c3aS, c3aD, ei, asrc, adst, emax, deno,
              out, NC, /*concat=*/0, 1.0f / 6.0f, N, E, ET, 6, 121, stream);
}
